// Hypernet_27058293964931
// MI455X (gfx1250) — matrix operand order repaired and checked by reading, not compiled, not run on hardware
//
#include <hip/hip_runtime.h>
#include <hip/hip_fp16.h>
#include <math.h>

#define H      256
#define TIN    128
#define TOUT   1680
#define NSAMP  262144
#define NWG    8          // LSTM workgroups
#define RPW    128        // gate rows per WG (4 gates * 32 hidden)
#define HPW    32         // hidden units per WG

typedef _Float16 v16h __attribute__((ext_vector_type(16)));
typedef float    v8f  __attribute__((ext_vector_type(8)));

union H16 { v16h v; _Float16 e[16]; };
union F8  { v8f  v; float    e[8];  };

// ---- workspace layout (bytes) ----
#define WS_CNT    0                    // int counter for grid barrier
#define WS_H      64                   // 256 f32: current h vector
#define WS_WPART  2048                 // 1680*8 f32: per-WG partial fc1 dots

#if __has_builtin(__builtin_amdgcn_sched_barrier)
#  define SCHED_BARRIER() __builtin_amdgcn_sched_barrier(0)
#else
#  define SCHED_BARRIER()
#endif

__device__ __forceinline__ float fexp2(float x) { return __builtin_amdgcn_exp2f(x); }
__device__ __forceinline__ float frcp(float x)  { return __builtin_amdgcn_rcpf(x); }
__device__ __forceinline__ float sigm(float x)  { return frcp(1.f + fexp2(-1.44269504f * x)); }
__device__ __forceinline__ float ftanh(float x) {
#if __has_builtin(__builtin_amdgcn_tanhf)
    return __builtin_amdgcn_tanhf(x);
#elif __has_builtin(__builtin_amdgcn_tanh_f32)
    return __builtin_amdgcn_tanh_f32(x);
#else
    float e = fexp2(2.88539008f * x);       // exp(2x)
    return (e - 1.f) * frcp(e + 1.f);
#endif
}

__global__ void k_init(int* cnt, float* hg) {
    if (blockIdx.x == 0 && threadIdx.x == 0) *cnt = 0;
    int t = blockIdx.x * blockDim.x + threadIdx.x;
    if (t < H) hg[t] = 0.f;
}

__device__ __forceinline__ void grid_sync(int* cnt, int target) {
    __threadfence();
    __syncthreads();
    if (threadIdx.x == 0) {
        atomicAdd(cnt, 1);
        while (__hip_atomic_load(cnt, __ATOMIC_ACQUIRE, __HIP_MEMORY_SCOPE_AGENT) < target)
            __builtin_amdgcn_s_sleep(1);
    }
    __syncthreads();
}

__global__ void __launch_bounds__(RPW, 1) k_lstm(
    const float* __restrict__ p,
    const float* __restrict__ W_ih1,
    const float* __restrict__ b_ih1, const float* __restrict__ b_hh1,
    const float* __restrict__ W_hh1,
    const float* __restrict__ W_ih2,
    const float* __restrict__ b_ih2, const float* __restrict__ b_hh2,
    const float* __restrict__ W_hh2,
    const float* __restrict__ fc1,
    int* cnt, float* hg, float* wpart) {

    __shared__ _Float16 h16[H];        // current h, f16, fed to WMMA B
    __shared__ float gbuf[RPW];
    __shared__ float cvec[RPW];
    __shared__ float wih[RPW];
    __shared__ float bl[RPW];
    __shared__ float fcl[HPW];
    __shared__ float red[HPW];

    const int s    = threadIdx.x;      // 0..127 : local gate row
    const int w    = blockIdx.x;       // 0..7
    const int lane = s & 31;
    const int wv   = s >> 5;           // wave id == gate block id
    const int m    = lane & 15;
    const int hi   = lane >> 4;
    const int r_s  = (s >> 5) * H + w * HPW + (s & 31);   // this thread's gate row

    wih[s] = W_ih1[r_s];
    bl[s]  = b_ih1[r_s] + b_hh1[r_s];
    if (s < HPW) fcl[s] = fc1[w * HPW + s];
    h16[s] = (_Float16)0.f; h16[s + RPW] = (_Float16)0.f;
    __syncthreads();

    // ---- load W_hh slice into WMMA A-fragments (register resident) --------
    // wave wv covers local rows [wv*32, wv*32+32) = 2 tiles of 16
    v16h fw[2][8];
    auto load_frags = [&](const float* Wm) {
        #pragma unroll
        for (int tt = 0; tt < 2; ++tt) {
            const int srow = wv * 32 + tt * 16 + m;                       // local row
            const int gr   = (srow >> 5) * H + w * HPW + (srow & 31);     // global row
            const float* wp = Wm + (size_t)gr * H;
            #pragma unroll
            for (int kc = 0; kc < 8; ++kc) {
                H16 tf;
                #pragma unroll
                for (int e = 0; e < 16; ++e) {
                    // contiguous K order (matches B): K = hi*16 + e
                    int K = kc * 32 + hi * 16 + e;
                    tf.e[e] = (_Float16)wp[K];
                }
                fw[tt][kc] = tf.v;
            }
        }
    };
    load_frags(W_hh1);

    float c = 0.f;
    int phase = 0;

    // ---------------- Stage 1: 128 steps, scalar input tanh(p[t]) ----------
    for (int t = 0; t < TIN; ++t) {
        float xt = ftanh(p[t]);
        // B fragments: B[k][n] = h[k] for all n (columns replicated)
        v16h bf[8];
        #pragma unroll
        for (int kc = 0; kc < 8; ++kc)
            bf[kc] = *(const v16h*)&h16[kc * 32 + hi * 16];
        SCHED_BARRIER();               // issue all ds_loads before the WMMA chain
        F8 c0, c1; v8f z = {}; c0.v = z; c1.v = z;
        #pragma unroll
        for (int kc = 0; kc < 8; ++kc) {
            c0.v = __builtin_amdgcn_wmma_f32_16x16x32_f16(false, fw[0][kc], false, bf[kc],
                                                          (short)0, c0.v, false, false);
            c1.v = __builtin_amdgcn_wmma_f32_16x16x32_f16(false, fw[1][kc], false, bf[kc],
                                                          (short)0, c1.v, false, false);
        }
        if (m == 0) {   // lanes 0 & 16 hold all 16 rows of each tile
            #pragma unroll
            for (int rr = 0; rr < 8; ++rr) {
                int row0 = wv * 32 + hi * 8 + rr;
                int row1 = row0 + 16;
                gbuf[row0] = c0.e[rr] + bl[row0] + xt * wih[row0];
                gbuf[row1] = c1.e[rr] + bl[row1] + xt * wih[row1];
            }
        }
        __syncthreads();
        if (s < HPW) {
            float gi = sigm(gbuf[s]);
            float gf = sigm(gbuf[32 + s]);
            float gg = ftanh(gbuf[64 + s]);
            float go = sigm(gbuf[96 + s]);
            c = gf * c + gi * gg;
            float h = go * ftanh(c);
            __hip_atomic_store(&hg[w * HPW + s], h, __ATOMIC_RELAXED, __HIP_MEMORY_SCOPE_AGENT);
        }
        ++phase;
        grid_sync(cnt, NWG * phase);
        h16[s]       = (_Float16)__hip_atomic_load(&hg[s],       __ATOMIC_RELAXED, __HIP_MEMORY_SCOPE_AGENT);
        h16[s + RPW] = (_Float16)__hip_atomic_load(&hg[s + RPW], __ATOMIC_RELAXED, __HIP_MEMORY_SCOPE_AGENT);
        __syncthreads();
    }

    // ------- Stage 2 precompute: cvec = b2 + W_ih2 @ h1 (step-invariant) ----
    {
        float acc = b_ih2[r_s] + b_hh2[r_s];
        const float* wp = W_ih2 + (size_t)r_s * H;
        #pragma unroll 4
        for (int j = 0; j < H; ++j)
            acc += wp[j] * __hip_atomic_load(&hg[j], __ATOMIC_RELAXED, __HIP_MEMORY_SCOPE_AGENT);
        cvec[s] = acc;
    }
    load_frags(W_hh2);                  // overwrite fragments with stage-2 weights
    __syncthreads();
    c = 0.f;
    h16[s] = (_Float16)0.f; h16[s + RPW] = (_Float16)0.f;  // stage-2 zero state
    __syncthreads();

    // ---------------- Stage 2: 1680 steps, emit h.fc1 per step --------------
    for (int t = 0; t < TOUT; ++t) {
        v16h bf[8];
        #pragma unroll
        for (int kc = 0; kc < 8; ++kc)
            bf[kc] = *(const v16h*)&h16[kc * 32 + hi * 16];
        SCHED_BARRIER();               // issue all ds_loads before the WMMA chain
        F8 c0, c1; v8f z = {}; c0.v = z; c1.v = z;
        #pragma unroll
        for (int kc = 0; kc < 8; ++kc) {
            c0.v = __builtin_amdgcn_wmma_f32_16x16x32_f16(false, fw[0][kc], false, bf[kc],
                                                          (short)0, c0.v, false, false);
            c1.v = __builtin_amdgcn_wmma_f32_16x16x32_f16(false, fw[1][kc], false, bf[kc],
                                                          (short)0, c1.v, false, false);
        }
        if (m == 0) {
            #pragma unroll
            for (int rr = 0; rr < 8; ++rr) {
                int row0 = wv * 32 + hi * 8 + rr;
                int row1 = row0 + 16;
                gbuf[row0] = c0.e[rr] + cvec[row0];
                gbuf[row1] = c1.e[rr] + cvec[row1];
            }
        }
        __syncthreads();
        if (s < HPW) {
            float gi = sigm(gbuf[s]);
            float gf = sigm(gbuf[32 + s]);
            float gg = ftanh(gbuf[64 + s]);
            float go = sigm(gbuf[96 + s]);
            c = gf * c + gi * gg;
            float h = go * ftanh(c);
            __hip_atomic_store(&hg[w * HPW + s], h, __ATOMIC_RELAXED, __HIP_MEMORY_SCOPE_AGENT);
            red[s] = h * fcl[s];
        }
        __syncthreads();
        if (s == 0) {                   // deterministic per-WG partial of h.fc1
            float sum = 0.f;
            #pragma unroll
            for (int i = 0; i < HPW; ++i) sum += red[i];
            wpart[t * NWG + w] = sum;
        }
        ++phase;
        grid_sync(cnt, NWG * phase);
        h16[s]       = (_Float16)__hip_atomic_load(&hg[s],       __ATOMIC_RELAXED, __HIP_MEMORY_SCOPE_AGENT);
        h16[s + RPW] = (_Float16)__hip_atomic_load(&hg[s + RPW], __ATOMIC_RELAXED, __HIP_MEMORY_SCOPE_AGENT);
        __syncthreads();
    }
}

// ---------- Generated MLP over N samples: WMMA f32_16x16x32_f16 -------------
__global__ void __launch_bounds__(256) k_mlp(
    const float* __restrict__ x, const float* __restrict__ wpart,
    float* __restrict__ out) {

    __shared__ float wl[TOUT];          // w matrix, 40x42 row-major
    for (int i = threadIdx.x; i < TOUT; i += 256) {
        float s = 0.f;
        #pragma unroll
        for (int q = 0; q < NWG; ++q) s += wpart[i * NWG + q];
        wl[i] = s;
    }
    __syncthreads();

    const int lane = threadIdx.x & 31;
    const int wv   = threadIdx.x >> 5;
    const int m    = lane & 15;
    const int hi   = lane >> 4;

    // B[k][n] = Wmid[n][k] = w[n][1+k]; padded K 40->64, N 40->48
    H16 bfrag[2][3];
    #pragma unroll
    for (int kc = 0; kc < 2; ++kc)
        #pragma unroll
        for (int nt = 0; nt < 3; ++nt) {
            int n = nt * 16 + m;
            #pragma unroll
            for (int e = 0; e < 16; ++e) {
                int kk = kc * 32 + hi * 16 + e;
                float v = (n < 40 && kk < 40) ? wl[n * 42 + 1 + kk] : 0.f;
                bfrag[kc][nt].e[e] = (_Float16)v;
            }
        }
    float woutv[3];
    #pragma unroll
    for (int nt = 0; nt < 3; ++nt) {
        int n = nt * 16 + m;
        woutv[nt] = (n < 40) ? wl[n * 42 + 41] : 0.f;
    }
    float w0v[2][16];
    #pragma unroll
    for (int kc = 0; kc < 2; ++kc)
        #pragma unroll
        for (int e = 0; e < 16; ++e) {
            int K  = hi * 16 + e;
            int kk = kc * 32 + K;
            w0v[kc][e] = (kk < 40) ? wl[kk * 42] : 0.f;
        }

    const int ntile  = NSAMP / 16;
    const int stride = gridDim.x * 8;
    for (int tile = blockIdx.x * 8 + wv; tile < ntile; tile += stride) {
        float xv = x[tile * 16 + m];
        if (tile + stride < ntile)
            __builtin_prefetch(&x[(tile + stride) * 16 + m], 0, 0);

        H16 afrag[2];                   // A[m][k] = relu(x_m * w0[k])
        #pragma unroll
        for (int kc = 0; kc < 2; ++kc)
            #pragma unroll
            for (int e = 0; e < 16; ++e)
                afrag[kc].e[e] = (_Float16)fmaxf(xv * w0v[kc][e], 0.f);

        F8 acc[3];
        #pragma unroll
        for (int nt = 0; nt < 3; ++nt) {
            v8f z = {};
            z = __builtin_amdgcn_wmma_f32_16x16x32_f16(false, afrag[0].v, false, bfrag[0][nt].v,
                                                       (short)0, z, false, false);
            z = __builtin_amdgcn_wmma_f32_16x16x32_f16(false, afrag[1].v, false, bfrag[1][nt].v,
                                                       (short)0, z, false, false);
            acc[nt].v = z;
        }

        float pr[8];
        #pragma unroll
        for (int rr = 0; rr < 8; ++rr) {
            float sv = 0.f;
            #pragma unroll
            for (int nt = 0; nt < 3; ++nt)
                sv += fmaxf(acc[nt].e[rr], 0.f) * woutv[nt];
            #pragma unroll
            for (int off = 1; off < 16; off <<= 1)
                sv += __shfl_xor(sv, off, 32);
            pr[rr] = sv;
        }
        if (m == 0) {                   // lane 0 -> rows 0..7, lane 16 -> rows 8..15
            int base = tile * 16 + hi * 8;
            #pragma unroll
            for (int rr = 0; rr < 8; ++rr)
                out[base + rr] = fmaxf(pr[rr], 0.f);
        }
    }
}

extern "C" void kernel_launch(void* const* d_in, const int* in_sizes, int n_in,
                              void* d_out, int out_size, void* d_ws, size_t ws_size,
                              hipStream_t stream) {
    const float* p     = (const float*)d_in[0];
    const float* xin   = (const float*)d_in[1];
    const float* W_ih1 = (const float*)d_in[2];
    const float* W_hh1 = (const float*)d_in[3];
    const float* b_ih1 = (const float*)d_in[4];
    const float* b_hh1 = (const float*)d_in[5];
    const float* W_ih2 = (const float*)d_in[6];
    const float* W_hh2 = (const float*)d_in[7];
    const float* b_ih2 = (const float*)d_in[8];
    const float* b_hh2 = (const float*)d_in[9];
    const float* fc1   = (const float*)d_in[10];
    float* out = (float*)d_out;

    char* ws = (char*)d_ws;
    int*   cnt   = (int*)(ws + WS_CNT);
    float* hg    = (float*)(ws + WS_H);
    float* wpart = (float*)(ws + WS_WPART);

    k_init<<<1, 256, 0, stream>>>(cnt, hg);
    k_lstm<<<NWG, RPW, 0, stream>>>(p, W_ih1, b_ih1, b_hh1, W_hh1,
                                    W_ih2, b_ih2, b_hh2, W_hh2,
                                    fc1, cnt, hg, wpart);
    k_mlp<<<1024, 256, 0, stream>>>(xin, wpart, out);
}
